// Decoder_83099027243701
// MI455X (gfx1250) — matrix operand order repaired and checked by reading, not compiled, not run on hardware
//
#include <hip/hip_runtime.h>

#define PRED_LEN 12
#define HIDDEN   64
#define BATCH    131072

typedef __attribute__((ext_vector_type(16))) _Float16 v16h;
typedef __attribute__((ext_vector_type(8)))  _Float16 v8h;
typedef __attribute__((ext_vector_type(8)))  float    v8f;

#define LOG2E 1.44269504088896340736f

// gfx1250 has a hardware transcendental tanh (V_TANH_F32, TRANS32 class).
// Prefer it; fall back to exp2/rcp if this toolchain lacks the builtin.
__device__ __forceinline__ float ftanh(float x) {
#if __has_builtin(__builtin_amdgcn_tanhf)
    return __builtin_amdgcn_tanhf(x);
#else
    // tanh(x) = 1 - 2/(1 + e^{2x})
    return 1.0f - 2.0f * __builtin_amdgcn_rcpf(1.0f + __builtin_amdgcn_exp2f(2.0f * LOG2E * x));
#endif
}
__device__ __forceinline__ float fsigmoid(float x) {
#if __has_builtin(__builtin_amdgcn_tanhf)
    // sigmoid(x) = 0.5*tanh(x/2) + 0.5  (3 VALU ops with HW tanh)
    return fmaf(0.5f, __builtin_amdgcn_tanhf(0.5f * x), 0.5f);
#else
    return __builtin_amdgcn_rcpf(1.0f + __builtin_amdgcn_exp2f(-LOG2E * x));
#endif
}

// One wave of 32 lanes owns 16 batch rows. gates[16,256] per LSTM step is done
// as 16 N-tiles x 2 K-chunks of v_wmma_f32_16x16x32_f16 with the x/bias term
// pre-loaded into the accumulator in C-layout. W_hh lives in LDS in B-fragment
// layout; h is restaged through wave-private LDS into A-fragment layout each step.
__global__ __launch_bounds__(256)
void lstm_decoder_wmma(const float* __restrict__ enc_h,
                       const float* __restrict__ enc_c,
                       const float* __restrict__ obs_pos,
                       const float* __restrict__ obs_rel,
                       const float* __restrict__ W_emb,
                       const float* __restrict__ b_emb,
                       const float* __restrict__ W_ih,
                       const float* __restrict__ W_hh,
                       const float* __restrict__ b_ih,
                       const float* __restrict__ b_hh,
                       const float* __restrict__ W_pos,
                       const float* __restrict__ b_pos,
                       float* __restrict__ out)
{
    // 32 B-fragments (16 N-tiles x 2 K-chunks), 512 halves each = 32 KB
    __shared__ _Float16 sWhh[32 * 512];
    // per-wave h staging buffer, [16 rows][64 cols] f16 = 2 KB/wave
    __shared__ _Float16 sH[8][16 * 64];

    const int tid  = threadIdx.x;
    const int wave = tid >> 5;
    const int lane = tid & 31;
    const int g    = lane >> 4;   // lane half-group
    const int nl   = lane & 15;   // N (or M) coordinate within tile

    // ---- Stage W_hh (f32 -> f16) into LDS, pre-swizzled to B-fragment layout.
    // Fragment f = 2*j + kc (j = N-tile, kc = K-chunk). Lane l: n = l&15,
    // K half e -> k = kc*32 + (l>>4)*16 + e.  B[k,n] = W_hh[16j+n][k].
    #pragma unroll 4
    for (int idx = tid; idx < 32 * 512; idx += 256) {
        int f = idx >> 9;
        int l = (idx >> 4) & 31;
        int e = idx & 15;
        int j  = f >> 1;
        int kc = f & 1;
        int n  = l & 15;
        int k  = kc * 32 + (l >> 4) * 16 + e;
        sWhh[idx] = (_Float16)W_hh[(16 * j + n) * 64 + k];
    }
    __syncthreads();

    // This lane's 8 rows are rowBase..rowBase+7 (C/D layout: VGPR r -> m = r + 8g)
    const long rowBase = (long)blockIdx.x * 128 + wave * 16 + g * 8;

    // ---- Load h, c in WMMA C/D fragment layout: tile tt covers cols 16tt..16tt+15.
    v8f h4[4], c4[4];
    #pragma unroll
    for (int tt = 0; tt < 4; ++tt) {
        #pragma unroll
        for (int r = 0; r < 8; ++r) {
            h4[tt][r] = enc_h[(rowBase + r) * 64 + nl + 16 * tt];
            c4[tt][r] = enc_c[(rowBase + r) * 64 + nl + 16 * tt];
        }
    }

    // ---- Per-lane weight/bias preloads (indexed by this lane's n coordinate).
    float wih0[16], wih1[16], bsum[16];
    #pragma unroll
    for (int j = 0; j < 16; ++j) {
        int ng = 16 * j + nl;
        wih0[j] = W_ih[ng * 2 + 0];
        wih1[j] = W_ih[ng * 2 + 1];
        bsum[j] = b_ih[ng] + b_hh[ng];
    }
    float wp0[4], wp1[4];
    #pragma unroll
    for (int tt = 0; tt < 4; ++tt) {
        wp0[tt] = W_pos[     nl + 16 * tt];
        wp1[tt] = W_pos[64 + nl + 16 * tt];
    }
    const float bp0 = b_pos[0], bp1 = b_pos[1];
    const float we00 = W_emb[0], we01 = W_emb[1], we10 = W_emb[2], we11 = W_emb[3];
    const float be0 = b_emb[0],  be1 = b_emb[1];

    // ---- Per-row state: running position and embedded input x.
    float cp0[8], cp1[8], x0[8], x1[8];
    #pragma unroll
    for (int r = 0; r < 8; ++r) {
        long row = rowBase + r;
        cp0[r] = obs_pos[row * 2 + 0];
        cp1[r] = obs_pos[row * 2 + 1];
        float rx0 = obs_rel[row * 2 + 0];
        float rx1 = obs_rel[row * 2 + 1];
        x0[r] = fmaf(rx0, we00, fmaf(rx1, we01, be0));
        x1[r] = fmaf(rx0, we10, fmaf(rx1, we11, be1));
    }

    _Float16* hb = sH[wave];

    for (int t = 0; t < PRED_LEN; ++t) {
        // -- Restage h (f16) to wave-private LDS [16][64], row-major.
        #pragma unroll
        for (int tt = 0; tt < 4; ++tt)
            #pragma unroll
            for (int r = 0; r < 8; ++r)
                hb[(g * 8 + r) * 64 + nl + 16 * tt] = (_Float16)h4[tt][r];
        asm volatile("s_wait_dscnt 0" ::: "memory");

        // -- Build A fragments (16x32 f16). A layout: row m = lane&15;
        //    halves e -> k = g*16+e (contiguous, matches sWhh B layout): two b128 runs.
        v16h A[2];
        #pragma unroll
        for (int a = 0; a < 2; ++a) {
            union { v16h v; v8h h2[2]; } u;
            u.h2[0] = *(const v8h*)&hb[nl * 64 + a * 32 + g * 16];
            u.h2[1] = *(const v8h*)&hb[nl * 64 + a * 32 + g * 16 + 8];
            A[a] = u.v;
        }

        // -- Gates per hidden chunk tt: tiles {tt, 4+tt, 8+tt, 12+tt} = i,f,g,o.
        #pragma unroll
        for (int tt = 0; tt < 4; ++tt) {
            v8f gt[4];
            #pragma unroll
            for (int q = 0; q < 4; ++q) {
                int j = 4 * q + tt;
                // C init = x @ W_ih^T + b_ih + b_hh in C-layout (2 FMAs/elem)
                v8f acc;
                #pragma unroll
                for (int r = 0; r < 8; ++r)
                    acc[r] = fmaf(x0[r], wih0[j], fmaf(x1[r], wih1[j], bsum[j]));
                const v16h B0 = *(const v16h*)&sWhh[(2 * j + 0) * 512 + lane * 16];
                const v16h B1 = *(const v16h*)&sWhh[(2 * j + 1) * 512 + lane * 16];
                acc = __builtin_amdgcn_wmma_f32_16x16x32_f16(false, A[0], false, B0,
                                                             (short)0, acc, false, false);
                acc = __builtin_amdgcn_wmma_f32_16x16x32_f16(false, A[1], false, B1,
                                                             (short)0, acc, false, false);
                gt[q] = acc;
            }
            #pragma unroll
            for (int r = 0; r < 8; ++r) {
                float iv = fsigmoid(gt[0][r]);
                float fv = fsigmoid(gt[1][r]);
                float gv = ftanh(gt[2][r]);
                float ov = fsigmoid(gt[3][r]);
                float cv = fmaf(fv, c4[tt][r], iv * gv);
                c4[tt][r] = cv;
                h4[tt][r] = ov * ftanh(cv);
            }
        }

        // -- rel_pos = h @ W_pos^T + b_pos (64 -> 2): per-lane partials over the
        //    4 n-values this lane holds, then butterfly-reduce across the 16-lane
        //    half-wave (xor masks 1,2,4,8 stay within each half).
        float p0[8], p1[8];
        #pragma unroll
        for (int r = 0; r < 8; ++r) {
            p0[r] = fmaf(h4[0][r], wp0[0], fmaf(h4[1][r], wp0[1],
                    fmaf(h4[2][r], wp0[2], h4[3][r] * wp0[3])));
            p1[r] = fmaf(h4[0][r], wp1[0], fmaf(h4[1][r], wp1[1],
                    fmaf(h4[2][r], wp1[2], h4[3][r] * wp1[3])));
        }
        #pragma unroll
        for (int m = 1; m < 16; m <<= 1) {
            #pragma unroll
            for (int r = 0; r < 8; ++r) {
                p0[r] += __shfl_xor(p0[r], m, 32);
                p1[r] += __shfl_xor(p1[r], m, 32);
            }
        }
        #pragma unroll
        for (int r = 0; r < 8; ++r) {
            p0[r] += bp0;  p1[r] += bp1;
            cp0[r] += p0[r];  cp1[r] += p1[r];
            // next x = rel_pos @ W_emb^T + b_emb
            x0[r] = fmaf(p0[r], we00, fmaf(p1[r], we01, be0));
            x1[r] = fmaf(p0[r], we10, fmaf(p1[r], we11, be1));
        }

        // -- Store: lane with nl<8 owns row rowBase+nl (all lanes in the half
        //    hold identical reduced values; select index nl).
        if (nl < 8) {
            float o0 = cp0[0], o1 = cp1[0], s0 = p0[0], s1 = p1[0];
            #pragma unroll
            for (int q = 1; q < 8; ++q)
                if (nl == q) { o0 = cp0[q]; o1 = cp1[q]; s0 = p0[q]; s1 = p1[q]; }
            long row = rowBase + nl;
            float2* o2 = (float2*)out;
            o2[(long)t * BATCH + row] = make_float2(o0, o1);                           // pred_seq
            o2[(long)PRED_LEN * BATCH + (long)t * BATCH + row] = make_float2(s0, s1);  // pred_seq_rel
        }
    }
}

extern "C" void kernel_launch(void* const* d_in, const int* in_sizes, int n_in,
                              void* d_out, int out_size, void* d_ws, size_t ws_size,
                              hipStream_t stream) {
    const float* enc_h   = (const float*)d_in[0];
    const float* enc_c   = (const float*)d_in[1];
    const float* obs_pos = (const float*)d_in[2];
    const float* obs_rel = (const float*)d_in[3];
    const float* W_emb   = (const float*)d_in[4];
    const float* b_emb   = (const float*)d_in[5];
    const float* W_ih    = (const float*)d_in[6];
    const float* W_hh    = (const float*)d_in[7];
    const float* b_ih    = (const float*)d_in[8];
    const float* b_hh    = (const float*)d_in[9];
    const float* W_pos   = (const float*)d_in[10];
    const float* b_pos   = (const float*)d_in[11];
    float* out = (float*)d_out;

    dim3 grid(BATCH / 128);   // 8 waves/block x 16 rows/wave
    dim3 block(256);
    lstm_decoder_wmma<<<grid, block, 0, stream>>>(
        enc_h, enc_c, obs_pos, obs_rel, W_emb, b_emb,
        W_ih, W_hh, b_ih, b_hh, W_pos, b_pos, out);
}
